// PrototypicalLearning_73882027426280
// MI455X (gfx1250) — matrix operand order repaired and checked by reading, not compiled, not run on hardware
//
#include <hip/hip_runtime.h>
#include <hip/hip_bf16.h>
#include <math.h>

// Problem constants (match reference)
#define NS 32768
#define NQ 16384
#define DD 512
#define KK 32
#define UU 128
#define NEG_BIG (-1e30f)

typedef __attribute__((ext_vector_type(16))) __bf16 v16bf;
typedef __attribute__((ext_vector_type(8)))  float  v8f;
typedef __attribute__((ext_vector_type(2)))  float  v2f;

union BfFrag { uint4 u[2]; v16bf v; };

__device__ __forceinline__ unsigned short f2bf(float f) {
    unsigned int x = __float_as_uint(f);
    unsigned int r = x + 0x7fffu + ((x >> 16) & 1u);   // round-to-nearest-even
    return (unsigned short)(r >> 16);
}

// ---------------------------------------------------------------------------
// Kernel 1: fp32 -> bf16 conversion of support (NS*DD) and V (KK*UU*DD)
// 8 elements per thread, 16B stores.
// ---------------------------------------------------------------------------
__global__ __launch_bounds__(256) void convert_bf16_kernel(
    const float* __restrict__ S, const float* __restrict__ V,
    unsigned short* __restrict__ Sb, unsigned short* __restrict__ Vb)
{
    size_t i = ((size_t)blockIdx.x * 256 + threadIdx.x) * 8;
    const size_t nS = (size_t)NS * DD;
    const float* src; unsigned short* dst; size_t j;
    if (i < nS) { src = S; dst = Sb; j = i; }
    else        { src = V; dst = Vb; j = i - nS; }
    float4 f0 = *(const float4*)(src + j);
    float4 f1 = *(const float4*)(src + j + 4);
    uint4 o;
    o.x = (unsigned)f2bf(f0.x) | ((unsigned)f2bf(f0.y) << 16);
    o.y = (unsigned)f2bf(f0.z) | ((unsigned)f2bf(f0.w) << 16);
    o.z = (unsigned)f2bf(f1.x) | ((unsigned)f2bf(f1.y) << 16);
    o.w = (unsigned)f2bf(f1.z) | ((unsigned)f2bf(f1.w) << 16);
    *(uint4*)(dst + j) = o;
}

// ---------------------------------------------------------------------------
// Kernel 2: attention logits via WMMA bf16.
// Grid: (NS/256, KK). Block: 256 threads = 8 waves. Each wave computes TWO
// 16-row M-tiles (32 support rows) x 128 u-columns, reusing every B fragment
// for two WMMAs (halves B traffic per FLOP). proj = tanh(support @ V[k]^T),
// logit = proj . w[k], reduced across lanes, written unmasked to logits[k][n].
// ---------------------------------------------------------------------------
__global__ __launch_bounds__(256) void logits_kernel(
    const unsigned short* __restrict__ Sb_, const unsigned short* __restrict__ Vb_,
    const float* __restrict__ w, float* __restrict__ logits)
{
    const __bf16* Sb = (const __bf16*)Sb_;
    const __bf16* Vb = (const __bf16*)Vb_;
    const int lane = threadIdx.x & 31;
    const int wave = threadIdx.x >> 5;
    const int l15  = lane & 15;
    const int half = lane >> 4;
    const int k    = blockIdx.y;
    const int row0 = blockIdx.x * 256 + wave * 32;   // first of 32 rows

    v8f acc0[8] = {};   // rows row0 .. row0+15
    v8f acc1[8] = {};   // rows row0+16 .. row0+31

    // A fragment (16x32 bf16): lane L(0-15)=row M=L holds K=0..7 (elems0-7) and
    // K=16..23 (elems8-15); lane L+16 holds K=8..15 / K=24..31.
    const __bf16* aBase0 = Sb + (size_t)(row0 + l15) * DD + half * 8;
    const __bf16* aBase1 = aBase0 + (size_t)16 * DD;
    // B fragment (32x16 bf16): lanes 0-15 = col n hold K=0..7 and K=16..23;
    // lanes 16-31 hold K=8..15 / K=24..31 (same order as A).
    const __bf16* bBase = Vb + ((size_t)k * UU + l15) * DD + half * 8;

    for (int d0 = 0; d0 < DD; d0 += 32) {
        BfFrag a0, a1;
        a0.u[0] = *(const uint4*)(aBase0 + d0);        // K chunk 0..7 (+half*8)
        a0.u[1] = *(const uint4*)(aBase0 + d0 + 16);   // K chunk 16..23 (+half*8)
        a1.u[0] = *(const uint4*)(aBase1 + d0);
        a1.u[1] = *(const uint4*)(aBase1 + d0 + 16);
#pragma unroll
        for (int t = 0; t < 8; ++t) {
            const __bf16* bp = bBase + (size_t)t * 16 * DD + d0;
            BfFrag b;
            b.u[0] = *(const uint4*)(bp);
            b.u[1] = *(const uint4*)(bp + 16);
            acc0[t] = __builtin_amdgcn_wmma_f32_16x16x32_bf16(
                false, a0.v, false, b.v, (short)0, acc0[t], false, false);
            acc1[t] = __builtin_amdgcn_wmma_f32_16x16x32_bf16(
                false, a1.v, false, b.v, (short)0, acc1[t], false, false);
        }
    }

    // Epilogue: partial[r] = sum over this lane's 8 u-columns of tanh(p)*w[u].
    float p0[8], p1[8];
#pragma unroll
    for (int r = 0; r < 8; ++r) { p0[r] = 0.0f; p1[r] = 0.0f; }
#pragma unroll
    for (int t = 0; t < 8; ++t) {
        float wt = w[k * UU + t * 16 + l15];
#pragma unroll
        for (int r = 0; r < 8; ++r) {
            p0[r] += tanhf(acc0[t][r]) * wt;
            p1[r] += tanhf(acc1[t][r]) * wt;
        }
    }
    // Reduce across the 16 lanes of each half (rows r in half0, r+8 in half1).
#pragma unroll
    for (int r = 0; r < 8; ++r) {
        float a = p0[r];
        a += __shfl_xor(a, 1, 32);
        a += __shfl_xor(a, 2, 32);
        a += __shfl_xor(a, 4, 32);
        a += __shfl_xor(a, 8, 32);
        float b = p1[r];
        b += __shfl_xor(b, 1, 32);
        b += __shfl_xor(b, 2, 32);
        b += __shfl_xor(b, 4, 32);
        b += __shfl_xor(b, 8, 32);
        if (l15 == 0) {
            logits[(size_t)k * NS + row0 + r + half * 8]      = a;
            logits[(size_t)k * NS + row0 + 16 + r + half * 8] = b;
        }
    }
}

// ---------------------------------------------------------------------------
// Kernel 3: per-class masked softmax stats (max, 1/sum). Empty class: max stays
// NEG_BIG, every element contributes exp(0)=1 -> uniform weights (mean fallback).
// ---------------------------------------------------------------------------
__global__ __launch_bounds__(256) void stats_kernel(
    const float* __restrict__ logits, const int* __restrict__ labels,
    float* __restrict__ stats)
{
    const int k = blockIdx.x;
    __shared__ float red[256];
    float m = NEG_BIG;
    for (int n = threadIdx.x; n < NS; n += 256) {
        float wl = (labels[n] == k) ? logits[(size_t)k * NS + n] : NEG_BIG;
        m = fmaxf(m, wl);
    }
    red[threadIdx.x] = m; __syncthreads();
    for (int s = 128; s > 0; s >>= 1) {
        if (threadIdx.x < s) red[threadIdx.x] = fmaxf(red[threadIdx.x], red[threadIdx.x + s]);
        __syncthreads();
    }
    m = red[0]; __syncthreads();
    float sum = 0.0f;
    for (int n = threadIdx.x; n < NS; n += 256) {
        float wl = (labels[n] == k) ? logits[(size_t)k * NS + n] : NEG_BIG;
        sum += expf(wl - m);
    }
    red[threadIdx.x] = sum; __syncthreads();
    for (int s = 128; s > 0; s >>= 1) {
        if (threadIdx.x < s) red[threadIdx.x] += red[threadIdx.x + s];
        __syncthreads();
    }
    if (threadIdx.x == 0) { stats[k * 2] = m; stats[k * 2 + 1] = 1.0f / red[0]; }
}

// ---------------------------------------------------------------------------
// Kernel 4: zero prototypes (d_ws is poisoned, must re-init every call).
// ---------------------------------------------------------------------------
__global__ void zero_protos_kernel(float* __restrict__ protos)
{
    int i = blockIdx.x * 256 + threadIdx.x;
    if (i < KK * DD) protos[i] = 0.0f;
}

// ---------------------------------------------------------------------------
// Kernel 5: prototypes[k][d] = sum_n attn[k][n] * support[n][d].
// Grid (16 row-chunks, KK). Rows with weight==0 skip the support read.
// ---------------------------------------------------------------------------
__global__ __launch_bounds__(256) void proto_kernel(
    const float* __restrict__ S, const int* __restrict__ labels,
    const float* __restrict__ logits, const float* __restrict__ stats,
    float* __restrict__ protos)
{
    const int k  = blockIdx.y;
    const int n0 = blockIdx.x * 2048;
    const float m = stats[k * 2], invs = stats[k * 2 + 1];
    const int d0 = threadIdx.x, d1 = threadIdx.x + 256;
    float a0 = 0.0f, a1 = 0.0f;
    for (int n = n0; n < n0 + 2048; ++n) {
        float wl  = (labels[n] == k) ? logits[(size_t)k * NS + n] : NEG_BIG;
        float wgt = expf(wl - m) * invs;
        if (wgt != 0.0f) {
            const float* row = S + (size_t)n * DD;
            a0 += wgt * row[d0];
            a1 += wgt * row[d1];
        }
    }
    atomicAdd(protos + k * DD + d0, a0);
    atomicAdd(protos + k * DD + d1, a1);
}

// ---------------------------------------------------------------------------
// Kernel 6: out = log_softmax(2*q.p - p2) over classes (q2 cancels in the
// log_softmax, so it is never computed). Prototypes staged in 64KB LDS.
// Tries the fp32 tensor op V_WMMA_F32_16X16X4_F32; VALU fallback otherwise.
// ---------------------------------------------------------------------------
#define HAVE_WMMA_F32X4 __has_builtin(__builtin_amdgcn_wmma_f32_16x16x4_f32)

__global__ __launch_bounds__(256) void out_kernel(
    const float* __restrict__ Q, const float* __restrict__ P,
    float* __restrict__ out)
{
    __shared__ float sP[KK * DD];                 // exactly 64 KB
    for (int i = threadIdx.x; i < KK * DD; i += 256) sP[i] = P[i];
    __syncthreads();

    const int lane = threadIdx.x & 31;
    const int wave = threadIdx.x >> 5;
    const int l15  = lane & 15;
    const int half = lane >> 4;
    const int q0   = blockIdx.x * 128 + wave * 16;

    // p2 for the two classes this lane owns in the C-tile layout.
    float p2a = 0.0f, p2b = 0.0f;
    for (int d = 0; d < DD; ++d) {
        float pa = sP[l15 * DD + d];
        float pb = sP[(l15 + 16) * DD + d];
        p2a += pa * pa; p2b += pb * pb;
    }

#if HAVE_WMMA_F32X4
    v8f acc0 = {}, acc1 = {};
    // A (16x4 f32): lane L(0-15)=row M=L, v0=K0,v1=K1; lane L+16: v0=K2,v1=K3.
    const float* qBase = Q + (size_t)(q0 + l15) * DD + half * 2;
    for (int d0 = 0; d0 < DD; d0 += 4) {
        float2 aq = *(const float2*)(qBase + d0);
        v2f a; a.x = aq.x; a.y = aq.y;
        // B (4x16 f32): lane n(0-15): v0=B[K=0][n], v1=B[K=1][n]; lanes 16-31: K=2,3.
        float2 b0q = *(const float2*)&sP[l15 * DD + d0 + half * 2];
        float2 b1q = *(const float2*)&sP[(l15 + 16) * DD + d0 + half * 2];
        v2f b0; b0.x = b0q.x; b0.y = b0q.y;
        v2f b1; b1.x = b1q.x; b1.y = b1q.y;
        acc0 = __builtin_amdgcn_wmma_f32_16x16x4_f32(
            false, a, false, b0, (short)0, acc0, false, false);
        acc1 = __builtin_amdgcn_wmma_f32_16x16x4_f32(
            false, a, false, b1, (short)0, acc1, false, false);
    }
#pragma unroll
    for (int r = 0; r < 8; ++r) {
        int m = r + half * 8;
        float l0 = 2.0f * acc0[r] - p2a;            // class l15
        float l1 = 2.0f * acc1[r] - p2b;            // class l15+16
        float mx = fmaxf(l0, l1);
        mx = fmaxf(mx, __shfl_xor(mx, 1, 32));
        mx = fmaxf(mx, __shfl_xor(mx, 2, 32));
        mx = fmaxf(mx, __shfl_xor(mx, 4, 32));
        mx = fmaxf(mx, __shfl_xor(mx, 8, 32));
        float s = expf(l0 - mx) + expf(l1 - mx);
        s += __shfl_xor(s, 1, 32);
        s += __shfl_xor(s, 2, 32);
        s += __shfl_xor(s, 4, 32);
        s += __shfl_xor(s, 8, 32);
        float lse = mx + logf(s);
        out[(size_t)(q0 + m) * KK + l15]      = l0 - lse;
        out[(size_t)(q0 + m) * KK + l15 + 16] = l1 - lse;
    }
#else
    // VALU fallback: threads 0..127 each handle one query of this block.
    if (threadIdx.x < 128) {
        int q = blockIdx.x * 128 + threadIdx.x;
        float lg[KK];
        for (int c = 0; c < KK; ++c) lg[c] = 0.0f;
        const float* qrow = Q + (size_t)q * DD;
        for (int d = 0; d < DD; ++d) {
            float qv = qrow[d];
            for (int c = 0; c < KK; ++c) lg[c] += qv * sP[c * DD + d];
        }
        float mx = NEG_BIG;
        for (int c = 0; c < KK; ++c) {
            float p2 = 0.0f;
            for (int d = 0; d < DD; ++d) p2 += sP[c * DD + d] * sP[c * DD + d];
            lg[c] = 2.0f * lg[c] - p2;
            mx = fmaxf(mx, lg[c]);
        }
        float s = 0.0f;
        for (int c = 0; c < KK; ++c) s += expf(lg[c] - mx);
        float lse = mx + logf(s);
        for (int c = 0; c < KK; ++c) out[(size_t)q * KK + c] = lg[c] - lse;
    }
#endif
}

// ---------------------------------------------------------------------------
extern "C" void kernel_launch(void* const* d_in, const int* in_sizes, int n_in,
                              void* d_out, int out_size, void* d_ws, size_t ws_size,
                              hipStream_t stream)
{
    const float* query   = (const float*)d_in[0];   // (NQ, DD)
    const float* support = (const float*)d_in[1];   // (NS, DD)
    const int*   labels  = (const int*)d_in[2];     // (NS,)
    const float* V       = (const float*)d_in[3];   // (KK, UU, DD)
    const float* w       = (const float*)d_in[4];   // (KK, UU)
    float* out = (float*)d_out;                     // (NQ, KK)

    char* ws = (char*)d_ws;
    const size_t nS = (size_t)NS * DD;              // support elems
    const size_t nV = (size_t)KK * UU * DD;         // V elems
    unsigned short* Sb  = (unsigned short*)(ws);                       // 32 MB
    unsigned short* Vb  = (unsigned short*)(ws + nS * 2);              //  4 MB
    float* logits = (float*)(ws + (nS + nV) * 2);                      //  4 MB
    float* stats  = (float*)(ws + (nS + nV) * 2 + (size_t)KK * NS * 4);
    float* protos = (float*)((char*)stats + 256);

    // 1) fp32 -> bf16
    {
        size_t total8 = (nS + nV) / 8;              // 2359296
        dim3 g((unsigned)(total8 / 256));           // 9216 blocks, exact
        convert_bf16_kernel<<<g, 256, 0, stream>>>(support, V, Sb, Vb);
    }
    // 2) attention logits (main WMMA GEMM), 32 rows per wave
    logits_kernel<<<dim3(NS / 256, KK), 256, 0, stream>>>(Sb, Vb, w, logits);
    // 3) softmax stats
    stats_kernel<<<KK, 256, 0, stream>>>(logits, labels, stats);
    // 4) zero + 5) accumulate prototypes
    zero_protos_kernel<<<(KK * DD + 255) / 256, 256, 0, stream>>>(protos);
    proto_kernel<<<dim3(16, KK), 256, 0, stream>>>(support, labels, logits, stats, protos);
    // 6) distances + log_softmax
    out_kernel<<<NQ / 128, 256, 0, stream>>>(query, protos, out);
}
